// VQEmbedding_71305047048235
// MI455X (gfx1250) — matrix operand order repaired and checked by reading, not compiled, not run on hardware
//
#include <hip/hip_runtime.h>
#include <hip/hip_bf16.h>

typedef __attribute__((ext_vector_type(16))) __bf16 v16bf;
typedef __attribute__((ext_vector_type(8)))  float  v8f;

#define KCODES 1024
#define DDIM   256
#define HW     1024            // H*W (d-stride of z in elements)
#define KC     256             // codes staged in LDS per chunk (128 KB bf16)
#define ROWS_PER_WAVE  32      // two 16-row A tiles per wave
#define ROWS_PER_BLOCK 256     // 8 waves * 32 rows

// ---------------------------------------------------------------------------
// Pre-pass: codebook fp32 -> bf16 (row-major; matches WMMA B-fragment layout)
// and per-code squared norms in fp32. One block (256 threads) per code row.
// ---------------------------------------------------------------------------
__global__ void __launch_bounds__(256)
vq_prep_kernel(const float* __restrict__ cb,
               __bf16* __restrict__ cb_bf,
               float* __restrict__ norms) {
    const int k = blockIdx.x;
    const int d = threadIdx.x;            // 0..255
    const float v = cb[(size_t)k * DDIM + d];
    cb_bf[(size_t)k * DDIM + d] = (__bf16)v;

    float s = v * v;
    #pragma unroll
    for (int off = 16; off > 0; off >>= 1)
        s += __shfl_down(s, off, 32);

    __shared__ float red[8];
    const int lane = threadIdx.x & 31;
    const int wv   = threadIdx.x >> 5;
    if (lane == 0) red[wv] = s;
    __syncthreads();
    if (threadIdx.x == 0) {
        float t = 0.f;
        #pragma unroll
        for (int i = 0; i < 8; ++i) t += red[i];
        norms[k] = t;
    }
}

// ---------------------------------------------------------------------------
// Main kernel: per wave, 32 z-rows (two 16-row WMMA A tiles) register-resident
// in bf16. Codebook streamed through LDS in 256-code chunks; each B fragment
// load feeds TWO v_wmma_f32_16x16x32_bf16 (one per A tile). B fragments are
// software-pipelined one deep so ds_loads for fragment c+1 are in flight
// while the WMMA pair for fragment c executes. Running argmin per lane,
// then 16-lane butterfly reduction.
// ---------------------------------------------------------------------------
__global__ void __launch_bounds__(256)
vq_argmin_kernel(const float* __restrict__ z,
                 const __bf16* __restrict__ cb_bf,
                 const float* __restrict__ norms,
                 float* __restrict__ out) {
    extern __shared__ char smem[];
    __bf16* lds_cb   = (__bf16*)smem;                           // KC * DDIM bf16
    float*  lds_norm = (float*)(smem + (size_t)KC * DDIM * 2);  // KC floats

    const int tid  = threadIdx.x;
    const int lane = tid & 31;
    const int wv   = tid >> 5;
    const int m    = lane & 15;     // row within tile (A) / code within k-tile (B,C)
    const int hi   = lane >> 4;     // lane-half selector per ISA fragment layouts

    const int n0 = blockIdx.x * ROWS_PER_BLOCK + wv * ROWS_PER_WAVE;
    const int nA = n0 + m;          // lane's row in tile 0
    const int nB = n0 + 16 + m;     // lane's row in tile 1
    const int bA = nA >> 10;        // n / (H*W)
    const int bB = nB >> 10;
    // z[b, d, h, w] flat = n + b*(D-1)*HW + d*HW
    const float* zrowA = z + (size_t)nA + (size_t)bA * ((DDIM - 1) * HW);
    const float* zrowB = z + (size_t)nB + (size_t)bB * ((DDIM - 1) * HW);

    // ---- Load A fragments once: 8 chunks of 16x32 bf16 per tile, contiguous k order (B matches)
    // lanes 0-15 (hi=0): K = c*32 + {0..15}
    // lanes 16-31 (hi=1): K = c*32 + {16..31}
    v16bf A0[8], A1[8];
    #pragma unroll
    for (int c = 0; c < 8; ++c) {
        const int dbase = c * 32 + hi * 16;
        #pragma unroll
        for (int i = 0; i < 8; ++i) {
            A0[c][i]     = (__bf16)zrowA[(size_t)(dbase + i)      * HW];
            A0[c][i + 8] = (__bf16)zrowA[(size_t)(dbase + 8 + i) * HW];
            A1[c][i]     = (__bf16)zrowB[(size_t)(dbase + i)      * HW];
            A1[c][i + 8] = (__bf16)zrowB[(size_t)(dbase + 8 + i) * HW];
        }
    }

    float bv0[8], bv1[8];
    int   bk0[8], bk1[8];
    #pragma unroll
    for (int r = 0; r < 8; ++r) {
        bv0[r] = 3.4e38f; bk0[r] = 0;
        bv1[r] = 3.4e38f; bk1[r] = 0;
    }

    for (int kc = 0; kc < KCODES; kc += KC) {
        // ---- Stage codebook chunk (128 KB bf16) + norms into LDS ----
        const uint4* src = (const uint4*)(cb_bf + (size_t)kc * DDIM);
        uint4*       dst = (uint4*)lds_cb;
        #pragma unroll 4
        for (int u = tid; u < (KC * DDIM * 2) / 16; u += 256)
            dst[u] = src[u];
        if (tid < KC) lds_norm[tid] = norms[kc + tid];

        // Hint-prefetch next chunk while we compute on this one.
        if (kc + KC < KCODES)
            __builtin_prefetch(cb_bf + (size_t)(kc + KC) * DDIM + (size_t)tid * 512, 0, 1);

        __syncthreads();

        #pragma unroll 1
        for (int t = 0; t < KC / 16; ++t) {
            const int kk = t * 16 + m;  // lane's code within chunk
            // B fragment (32x16 bf16): lane = code column, d = c*32 + hi*16 + 0..15
            const v16bf* bp =
                (const v16bf*)(lds_cb + (size_t)kk * DDIM + hi * 16);

            // Issue norm load + first B fragment load up front; WMMAs for
            // fragment c overlap the ds_loads of fragment c+1.
            const float en = lds_norm[kk];
            v16bf Bcur = bp[0];

            v8f acc0 = {0.f, 0.f, 0.f, 0.f, 0.f, 0.f, 0.f, 0.f};
            v8f acc1 = {0.f, 0.f, 0.f, 0.f, 0.f, 0.f, 0.f, 0.f};
            #pragma unroll
            for (int c = 0; c < 8; ++c) {
                v16bf Bnext;
                if (c < 7) Bnext = bp[(c + 1) * 2];  // +32 bf16 elements per step
                acc0 = __builtin_amdgcn_wmma_f32_16x16x32_bf16(
                    false, A0[c], false, Bcur, (short)0, acc0, false, false);
                acc1 = __builtin_amdgcn_wmma_f32_16x16x32_bf16(
                    false, A1[c], false, Bcur, (short)0, acc1, false, false);
                if (c < 7) Bcur = Bnext;
            }

            const int kg = kc + kk;
            #pragma unroll
            for (int r = 0; r < 8; ++r) {
                const float s0 = fmaf(-2.0f, acc0[r], en);  // ||e||^2 - 2 z.e
                if (s0 < bv0[r]) { bv0[r] = s0; bk0[r] = kg; }
                const float s1 = fmaf(-2.0f, acc1[r], en);
                if (s1 < bv1[r]) { bv1[r] = s1; bk1[r] = kg; }
            }
        }
        __syncthreads();
    }

    // ---- Reduce argmin across the 16 lanes sharing each row set ----
    // C/D layout: lane holds rows M = hi*8 + r, code N = m.
    #pragma unroll
    for (int r = 0; r < 8; ++r) {
        float v0 = bv0[r]; int k0 = bk0[r];
        float v1 = bv1[r]; int k1 = bk1[r];
        #pragma unroll
        for (int off = 1; off < 16; off <<= 1) {
            const float ov0 = __shfl_xor(v0, off, 32);
            const int   ok0 = __shfl_xor(k0, off, 32);
            if (ov0 < v0 || (ov0 == v0 && ok0 < k0)) { v0 = ov0; k0 = ok0; }
            const float ov1 = __shfl_xor(v1, off, 32);
            const int   ok1 = __shfl_xor(k1, off, 32);
            if (ov1 < v1 || (ov1 == v1 && ok1 < k1)) { v1 = ov1; k1 = ok1; }
        }
        if (m == 0) {
            out[n0 + hi * 8 + r]      = (float)k0;  // tile 0 rows
            out[n0 + 16 + hi * 8 + r] = (float)k1;  // tile 1 rows
        }
    }
}

// ---------------------------------------------------------------------------
extern "C" void kernel_launch(void* const* d_in, const int* in_sizes, int n_in,
                              void* d_out, int out_size, void* d_ws, size_t ws_size,
                              hipStream_t stream) {
    const float* z  = (const float*)d_in[0];   // [64, 256, 32, 32] fp32
    const float* cb = (const float*)d_in[1];   // [1024, 256]       fp32

    __bf16* cb_bf = (__bf16*)d_ws;                                     // 512 KB
    float*  norms = (float*)((char*)d_ws + (size_t)KCODES * DDIM * 2); // 4 KB
    float*  out   = (float*)d_out;                                     // 65536 indices

    vq_prep_kernel<<<KCODES, 256, 0, stream>>>(cb, cb_bf, norms);

    const int n_rows = 64 * 32 * 32;                 // 65536
    const int grid   = n_rows / ROWS_PER_BLOCK;      // 256 blocks
    const size_t shmem = (size_t)KC * DDIM * 2 + (size_t)KC * 4;  // 132 KB
    vq_argmin_kernel<<<grid, 256, shmem, stream>>>(z, cb_bf, norms, out);
}
